// Seq2SeqAttentionDecoder_15960098472740
// MI455X (gfx1250) — compile-verified, matrix operand order checked by reading, not run on hardware
//
#include <hip/hip_runtime.h>
#include <hip/hip_bf16.h>

// Shapes from the reference
#define V_SZ 10000
#define E_SZ 256
#define H_SZ 512
#define B_SZ 64
#define T_SZ 64
#define S_SZ 128

typedef __attribute__((ext_vector_type(16))) __bf16         bf16x16;
typedef __attribute__((ext_vector_type(8)))  float          floatx8;
typedef __attribute__((ext_vector_type(8)))  unsigned short ushort8;
typedef __attribute__((ext_vector_type(4)))  int            i32x4;

#if defined(__AMDGCN__) && __has_builtin(__builtin_amdgcn_global_load_async_to_lds_b128) && __has_builtin(__builtin_amdgcn_s_wait_asynccnt)
#define HAVE_ASYNC_LDS 1
// Builtin signature (from hipcc diagnostic): (v4i AS1*, v4i AS3*, imm offset, imm cpol)
typedef __attribute__((address_space(1))) i32x4 gl_i32x4;
typedef __attribute__((address_space(3))) i32x4 lds_i32x4;
#else
#define HAVE_ASYNC_LDS 0
#endif

// ---------------- helpers ----------------

__device__ __forceinline__ unsigned short f2bf(float f) {
  unsigned u = __float_as_uint(f);
  unsigned r = 0x7FFFu + ((u >> 16) & 1u);   // round-to-nearest-even
  return (unsigned short)((u + r) >> 16);
}

// Load a 16x32 bf16 WMMA A/B fragment for this lane from global memory.
// ISA layout (16-bit operand, wave32): lanes 0-15 -> rows 0-15 holding K
// {0..7,16..23}; lanes 16-31 -> rows 0-15 holding K {8..15,24..31}.
__device__ __forceinline__ bf16x16 load_frag(const unsigned short* __restrict__ base,
                                             int row, int ld, int kb, int lane) {
  const unsigned short* p = base + (size_t)row * ld + kb + ((lane >> 4) << 3);
  union { ushort8 h[2]; bf16x16 v; } u;
  u.h[0] = *(const ushort8*)(p);
  u.h[1] = *(const ushort8*)(p + 16);
  return u.v;
}

// Same fragment pattern from an LDS-resident tile row (32 bf16 per row, ld=32).
__device__ __forceinline__ bf16x16 lds_frag(const unsigned short* p, int lane) {
  int hi = (lane >> 4) << 3;
  union { ushort8 h[2]; bf16x16 v; } u;
  u.h[0] = *(const ushort8*)(p + hi);
  u.h[1] = *(const ushort8*)(p + 16 + hi);
  return u.v;
}

__device__ __forceinline__ floatx8 wmma_bf16(bf16x16 a, bf16x16 b, floatx8 c) {
  return __builtin_amdgcn_wmma_f32_16x16x32_bf16(
      /*neg_a=*/false, a, /*neg_b=*/false, b,
      /*c_mod=*/(short)0, c, /*reuse_a=*/false, /*reuse_b=*/false);
}

// 16-byte Global -> LDS transfer: async copy (ASYNCcnt) when available,
// register bounce (global_load_b128 + ds_store_b128) otherwise.
__device__ __forceinline__ void stage16(const unsigned short* g, unsigned short* l) {
#if HAVE_ASYNC_LDS
  __builtin_amdgcn_global_load_async_to_lds_b128(
      (gl_i32x4*)(unsigned short*)g,   // strip const, then addrspacecast to AS1
      (lds_i32x4*)l,                   // addrspacecast generic->AS3
      0, 0);
#else
  *(ushort8*)l = *(const ushort8*)g;
#endif
}

__device__ __forceinline__ void wait_async_stages() {
#if HAVE_ASYNC_LDS
  __builtin_amdgcn_s_wait_asynccnt(0);
#endif
}

#define ZERO8 {0.f,0.f,0.f,0.f,0.f,0.f,0.f,0.f}

// ---------------- one-time prep kernels ----------------

__global__ void cvt_bf16(const float* __restrict__ s, unsigned short* __restrict__ d, int n) {
  for (int i = blockIdx.x * blockDim.x + threadIdx.x; i < n; i += gridDim.x * blockDim.x)
    d[i] = f2bf(s[i]);
}

// X_bf time-major: [t, b, e] = emb[x[b,t], e] in bf16
__global__ void gather_embed(const int* __restrict__ x, const float* __restrict__ emb,
                             unsigned short* __restrict__ Xbf) {
  const int n = T_SZ * B_SZ * E_SZ;
  for (int i = blockIdx.x * blockDim.x + threadIdx.x; i < n; i += gridDim.x * blockDim.x) {
    int e = i & (E_SZ - 1);
    int b = (i >> 8) & (B_SZ - 1);   // E=256
    int t = i >> 14;                 // B*E = 16384
    int tok = x[b * T_SZ + t];
    Xbf[i] = f2bf(emb[(size_t)tok * E_SZ + e]);
  }
}

__global__ void init_h(const float* __restrict__ h0in,
                       float* __restrict__ h0f, unsigned short* __restrict__ h0b,
                       float* __restrict__ h1f, unsigned short* __restrict__ h1b) {
  const int n = B_SZ * H_SZ;
  for (int i = blockIdx.x * blockDim.x + threadIdx.x; i < n; i += gridDim.x * blockDim.x) {
    float a = h0in[i];     h0f[i] = a; h0b[i] = f2bf(a);
    float c = h0in[n + i]; h1f[i] = c; h1b[i] = f2bf(c);
  }
}

// ---------------- key projection GEMM: kproj[bs,o] = enc[bs,:] . Wk[o,:] ----------------
// grid (BS/16=512, H/64=8), block 32 (one wave). 4 N-tiles per block for A reuse.
__global__ void kproj_gemm(const unsigned short* __restrict__ enc_bf,
                           const unsigned short* __restrict__ Wk_bf,
                           float* __restrict__ kproj) {
  int lane = threadIdx.x;
  int m0 = blockIdx.x * 16;
  int n0 = blockIdx.y * 64;
  floatx8 acc[4] = {ZERO8, ZERO8, ZERO8, ZERO8};
  int arow = m0 + (lane & 15);
  for (int kb = 0; kb < H_SZ; kb += 32) {
    bf16x16 a = load_frag(enc_bf, arow, H_SZ, kb, lane);
#pragma unroll
    for (int j = 0; j < 4; ++j) {
      bf16x16 w = load_frag(Wk_bf, n0 + j * 16 + (lane & 15), H_SZ, kb, lane);
      acc[j] = wmma_bf16(a, w, acc[j]);
    }
  }
#pragma unroll
  for (int j = 0; j < 4; ++j)
#pragma unroll
    for (int r = 0; r < 8; ++r) {
      int m = r + ((lane >> 4) << 3);
      int n = lane & 15;
      kproj[(size_t)(m0 + m) * H_SZ + n0 + j * 16 + n] = acc[j][r];
    }
}

// ---------------- per-step attention (pure VALU fp32) ----------------
// grid (B), block 256. Produces inp_bf[b, 0:768] = [context_bf16 | x_t_bf16]
__global__ void attention_step(const float* __restrict__ h1f, const float* __restrict__ kproj,
                               const float* __restrict__ enc, const float* __restrict__ Wq,
                               const float* __restrict__ wv, const int* __restrict__ valid,
                               const unsigned short* __restrict__ Xt,
                               unsigned short* __restrict__ inp) {
  int b = blockIdx.x, tid = threadIdx.x;
  __shared__ float sh_h[H_SZ];
  __shared__ float sh_q[H_SZ];
  __shared__ float sh_sc[S_SZ];
  __shared__ float sh_red[256];
  __shared__ float sh_mx, sh_isum;

  for (int i = tid; i < H_SZ; i += 256) sh_h[i] = h1f[(size_t)b * H_SZ + i];
  __syncthreads();

  // q_proj = h1[b] @ Wq^T   (512x512 dot-products, fp32)
  for (int o = tid; o < H_SZ; o += 256) {
    const float* wrow = Wq + (size_t)o * H_SZ;
    float s = 0.f;
#pragma unroll 8
    for (int h = 0; h < H_SZ; ++h) s += sh_h[h] * wrow[h];
    sh_q[o] = s;
  }
  __syncthreads();

  // scores[s] = sum_h tanh(q[h] + kproj[b,s,h]) * wv[h]; 2 threads per s
  {
    int si = tid & (S_SZ - 1);
    int half = tid >> 7;
    const float* kp = kproj + ((size_t)b * S_SZ + si) * H_SZ;
    float p = 0.f;
    int h0 = half * (H_SZ / 2);
#pragma unroll 4
    for (int h = h0; h < h0 + H_SZ / 2; ++h)
      p += tanhf(sh_q[h] + kp[h]) * wv[h];
    sh_red[tid] = p;
  }
  __syncthreads();
  if (tid < S_SZ) {
    float sc = sh_red[tid] + sh_red[tid + S_SZ];
    if (tid >= valid[b]) sc = -1e6f;
    sh_sc[tid] = sc;
  }
  __syncthreads();
  if (tid == 0) {
    float mx = sh_sc[0];
    for (int s = 1; s < S_SZ; ++s) mx = fmaxf(mx, sh_sc[s]);
    sh_mx = mx;
  }
  __syncthreads();
  if (tid < S_SZ) sh_sc[tid] = __expf(sh_sc[tid] - sh_mx);
  __syncthreads();
  if (tid == 0) {
    float s = 0.f;
    for (int i = 0; i < S_SZ; ++i) s += sh_sc[i];
    sh_isum = 1.f / s;
  }
  __syncthreads();
  if (tid < S_SZ) sh_sc[tid] *= sh_isum;
  __syncthreads();

  // context[o] = sum_s attn[s] * enc[b,s,o]
  for (int o = tid; o < H_SZ; o += 256) {
    const float* eb = enc + (size_t)b * S_SZ * H_SZ + o;
    float c = 0.f;
#pragma unroll 4
    for (int s = 0; s < S_SZ; ++s) c += sh_sc[s] * eb[(size_t)s * H_SZ];
    inp[b * (H_SZ + E_SZ) + o] = f2bf(c);
  }
  for (int e = tid; e < E_SZ; e += 256)
    inp[b * (H_SZ + E_SZ) + H_SZ + e] = Xt[b * E_SZ + e];
}

// ---------------- GRU cell (one layer) via WMMA bf16 ----------------
// grid (B/16=4, H/16=32), block 32. Each wave computes six 16x16 gate tiles
// (gi/gh for r,z,n at columns c0, c0+H, c0+2H) then fuses the gate math.
__global__ void gru_step(const unsigned short* __restrict__ Abf, int Kin,
                         const unsigned short* __restrict__ hbf,
                         const float* __restrict__ hf,
                         const unsigned short* __restrict__ Wih,
                         const unsigned short* __restrict__ Whh,
                         const float* __restrict__ bih, const float* __restrict__ bhh,
                         float* __restrict__ hnf, unsigned short* __restrict__ hnb,
                         unsigned short* __restrict__ ysb) {
  int lane = threadIdx.x;
  int b0 = blockIdx.x * 16;
  int c0 = blockIdx.y * 16;
  int arow = b0 + (lane & 15);

  floatx8 aI[3] = {ZERO8, ZERO8, ZERO8};
  for (int kb = 0; kb < Kin; kb += 32) {
    bf16x16 a = load_frag(Abf, arow, Kin, kb, lane);
#pragma unroll
    for (int g = 0; g < 3; ++g) {
      bf16x16 w = load_frag(Wih, g * H_SZ + c0 + (lane & 15), Kin, kb, lane);
      aI[g] = wmma_bf16(a, w, aI[g]);
    }
  }
  floatx8 aH[3] = {ZERO8, ZERO8, ZERO8};
  for (int kb = 0; kb < H_SZ; kb += 32) {
    bf16x16 a = load_frag(hbf, arow, H_SZ, kb, lane);
#pragma unroll
    for (int g = 0; g < 3; ++g) {
      bf16x16 w = load_frag(Whh, g * H_SZ + c0 + (lane & 15), H_SZ, kb, lane);
      aH[g] = wmma_bf16(a, w, aH[g]);
    }
  }

#pragma unroll
  for (int r = 0; r < 8; ++r) {
    int m = r + ((lane >> 4) << 3);
    int n = lane & 15;
    int bi = b0 + m;
    int o = c0 + n;
    float ir  = aI[0][r] + bih[o];
    float hr  = aH[0][r] + bhh[o];
    float iz  = aI[1][r] + bih[H_SZ + o];
    float hz  = aH[1][r] + bhh[H_SZ + o];
    float inn = aI[2][r] + bih[2 * H_SZ + o];
    float hn  = aH[2][r] + bhh[2 * H_SZ + o];
    float rg = 1.f / (1.f + __expf(-(ir + hr)));
    float zg = 1.f / (1.f + __expf(-(iz + hz)));
    float ng = tanhf(inn + rg * hn);
    float hp = hf[(size_t)bi * H_SZ + o];
    float hv = (1.f - zg) * ng + zg * hp;
    hnf[(size_t)bi * H_SZ + o] = hv;
    unsigned short hb = f2bf(hv);
    hnb[(size_t)bi * H_SZ + o] = hb;
    if (ysb) ysb[(size_t)bi * H_SZ + o] = hb;
  }
}

// ---------------- final logits GEMM ----------------
// out[b,t,v] = ys[t,b,:] . Wd[v,:] + bd[v]
// grid (TB/64=64, V/80=125), block 128 (4 waves). Each wave owns one 16-row
// M-tile; the five 16x32 Wd K-slices per step are staged into a double-
// buffered LDS ring with async Global->LDS copies and shared by all 4 waves.
#define NT 5

__device__ __forceinline__ void stage_tiles(unsigned short* dst,
                                            const unsigned short* __restrict__ W,
                                            int n0, int kb, int tid) {
  // NT tiles * 16 rows * 4 chunks(16B) = 320 transfers, 128 threads
  for (int i = tid; i < NT * 16 * 4; i += 128) {
    int j   = i >> 6;        // tile
    int rem = i & 63;
    int row = rem >> 2;      // row within tile
    int c   = rem & 3;       // 16B chunk within the 64B row slice
    stage16(W + (size_t)(n0 + j * 16 + row) * H_SZ + kb + c * 8,
            dst + (j * 16 + row) * 32 + c * 8);
  }
}

__global__ void __launch_bounds__(128) logits_gemm(const unsigned short* __restrict__ ys,
                                                   const unsigned short* __restrict__ Wd_bf,
                                                   const float* __restrict__ bd,
                                                   float* __restrict__ out) {
  int tid  = threadIdx.x;
  int lane = tid & 31;
  int wid  = tid >> 5;                 // wave 0..3
  int m0 = blockIdx.x * 64 + wid * 16; // row r = t*B + b
  int n0 = blockIdx.y * (NT * 16);

  __shared__ unsigned short sb[2][NT * 16 * 32];  // 2 x 5KB B-tile ring

  stage_tiles(sb[0], Wd_bf, n0, 0, tid);
  wait_async_stages();
  __syncthreads();

  floatx8 acc[NT] = {ZERO8, ZERO8, ZERO8, ZERO8, ZERO8};
  int arow = m0 + (lane & 15);
  for (int i = 0; i < H_SZ / 32; ++i) {
    int kb  = i * 32;
    int cur = i & 1;
    if (i + 1 < H_SZ / 32)
      stage_tiles(sb[cur ^ 1], Wd_bf, n0, kb + 32, tid);
    bf16x16 a = load_frag(ys, arow, H_SZ, kb, lane);
#pragma unroll
    for (int j = 0; j < NT; ++j) {
      bf16x16 w = lds_frag(&sb[cur][(j * 16 + (lane & 15)) * 32], lane);
      acc[j] = wmma_bf16(a, w, acc[j]);
    }
    wait_async_stages();
    __syncthreads();
  }

#pragma unroll
  for (int j = 0; j < NT; ++j)
#pragma unroll
    for (int r = 0; r < 8; ++r) {
      int m = r + ((lane >> 4) << 3);
      int n = lane & 15;
      int row = m0 + m;
      int t = row >> 6;    // B = 64
      int b = row & 63;
      int v = n0 + j * 16 + n;
      out[((size_t)b * T_SZ + t) * V_SZ + v] = acc[j][r] + bd[v];
    }
}

// ---------------- host orchestration ----------------

extern "C" void kernel_launch(void* const* d_in, const int* in_sizes, int n_in,
                              void* d_out, int out_size, void* d_ws, size_t ws_size,
                              hipStream_t stream) {
  const int*   x     = (const int*)d_in[0];
  const float* enc   = (const float*)d_in[1];
  const float* h0in  = (const float*)d_in[2];
  const int*   valid = (const int*)d_in[3];
  const float* emb   = (const float*)d_in[4];
  const float* Wq    = (const float*)d_in[5];
  const float* Wk    = (const float*)d_in[6];
  const float* wv    = (const float*)d_in[7];
  const float* Wih0  = (const float*)d_in[8];
  const float* Whh0  = (const float*)d_in[9];
  const float* bih0  = (const float*)d_in[10];
  const float* bhh0  = (const float*)d_in[11];
  const float* Wih1  = (const float*)d_in[12];
  const float* Whh1  = (const float*)d_in[13];
  const float* bih1  = (const float*)d_in[14];
  const float* bhh1  = (const float*)d_in[15];
  const float* Wd    = (const float*)d_in[16];
  const float* bd    = (const float*)d_in[17];
  float* out = (float*)d_out;

  char* ws = (char*)d_ws;
  size_t off = 0;
  auto take = [&](size_t bytes) -> char* {
    char* p = ws + off;
    off += (bytes + 255) & ~(size_t)255;
    return p;
  };

  unsigned short* Xbf     = (unsigned short*)take((size_t)T_SZ * B_SZ * E_SZ * 2);
  unsigned short* enc_bf  = (unsigned short*)take((size_t)B_SZ * S_SZ * H_SZ * 2);
  float*          kproj   = (float*)take((size_t)B_SZ * S_SZ * H_SZ * 4);
  unsigned short* Wk_bf   = (unsigned short*)take((size_t)H_SZ * H_SZ * 2);
  unsigned short* Wih0_bf = (unsigned short*)take((size_t)3 * H_SZ * (H_SZ + E_SZ) * 2);
  unsigned short* Whh0_bf = (unsigned short*)take((size_t)3 * H_SZ * H_SZ * 2);
  unsigned short* Wih1_bf = (unsigned short*)take((size_t)3 * H_SZ * H_SZ * 2);
  unsigned short* Whh1_bf = (unsigned short*)take((size_t)3 * H_SZ * H_SZ * 2);
  unsigned short* Wd_bf   = (unsigned short*)take((size_t)V_SZ * H_SZ * 2);
  unsigned short* ys_bf   = (unsigned short*)take((size_t)T_SZ * B_SZ * H_SZ * 2);
  float*          h0f[2]  = {(float*)take((size_t)B_SZ * H_SZ * 4), (float*)take((size_t)B_SZ * H_SZ * 4)};
  float*          h1f[2]  = {(float*)take((size_t)B_SZ * H_SZ * 4), (float*)take((size_t)B_SZ * H_SZ * 4)};
  unsigned short* h0b[2]  = {(unsigned short*)take((size_t)B_SZ * H_SZ * 2), (unsigned short*)take((size_t)B_SZ * H_SZ * 2)};
  unsigned short* h1b[2]  = {(unsigned short*)take((size_t)B_SZ * H_SZ * 2), (unsigned short*)take((size_t)B_SZ * H_SZ * 2)};
  unsigned short* inp_bf  = (unsigned short*)take((size_t)B_SZ * (H_SZ + E_SZ) * 2);
  (void)ws_size; (void)in_sizes; (void)n_in; (void)out_size;

  // one-time prep: gather embedding + bf16 weight conversions + h init
  gather_embed<<<1024, 256, 0, stream>>>(x, emb, Xbf);
  cvt_bf16<<<1024, 256, 0, stream>>>(enc,  enc_bf,  B_SZ * S_SZ * H_SZ);
  cvt_bf16<<<256,  256, 0, stream>>>(Wk,   Wk_bf,   H_SZ * H_SZ);
  cvt_bf16<<<1024, 256, 0, stream>>>(Wih0, Wih0_bf, 3 * H_SZ * (H_SZ + E_SZ));
  cvt_bf16<<<1024, 256, 0, stream>>>(Whh0, Whh0_bf, 3 * H_SZ * H_SZ);
  cvt_bf16<<<1024, 256, 0, stream>>>(Wih1, Wih1_bf, 3 * H_SZ * H_SZ);
  cvt_bf16<<<1024, 256, 0, stream>>>(Whh1, Whh1_bf, 3 * H_SZ * H_SZ);
  cvt_bf16<<<2048, 256, 0, stream>>>(Wd,   Wd_bf,   V_SZ * H_SZ);
  init_h<<<128, 256, 0, stream>>>(h0in, h0f[0], h0b[0], h1f[0], h1b[0]);

  // key projection (WMMA)
  kproj_gemm<<<dim3(B_SZ * S_SZ / 16, H_SZ / 64), 32, 0, stream>>>(enc_bf, Wk_bf, kproj);

  // sequential recurrence: 64 steps, ping-pong hidden state buffers
  for (int t = 0; t < T_SZ; ++t) {
    int p = t & 1, q = p ^ 1;
    attention_step<<<B_SZ, 256, 0, stream>>>(h1f[p], kproj, enc, Wq, wv, valid,
                                             Xbf + (size_t)t * B_SZ * E_SZ, inp_bf);
    gru_step<<<dim3(B_SZ / 16, H_SZ / 16), 32, 0, stream>>>(
        inp_bf, H_SZ + E_SZ, h0b[p], h0f[p], Wih0_bf, Whh0_bf, bih0, bhh0,
        h0f[q], h0b[q], (unsigned short*)nullptr);
    gru_step<<<dim3(B_SZ / 16, H_SZ / 16), 32, 0, stream>>>(
        h0b[q], H_SZ, h1b[p], h1f[p], Wih1_bf, Whh1_bf, bih1, bhh1,
        h1f[q], h1b[q], ys_bf + (size_t)t * B_SZ * H_SZ);
  }

  // final projection to vocabulary (WMMA + async LDS-staged B tiles)
  logits_gemm<<<dim3(T_SZ * B_SZ / 64, V_SZ / (NT * 16)), 128, 0, stream>>>(ys_bf, Wd_bf, bd, out);
}
